// GraphNet_79087527788652
// MI455X (gfx1250) — matrix operand order repaired and checked by reading, not compiled, not run on hardware
//
#include <hip/hip_runtime.h>

#define N_NODES 50000
#define N_EDGES 400000
#define DIN 32
#define DH 128
#define DOUT 16
#define NBLK 2

typedef __attribute__((ext_vector_type(16))) __bf16 v16bf;
typedef __attribute__((ext_vector_type(8)))  __bf16 v8bf;
typedef __attribute__((ext_vector_type(8)))  float  v8f;
typedef __attribute__((ext_vector_type(4)))  float  v4f;

static __device__ __forceinline__ v8f wmma_bf16(v16bf a, v16bf b, v8f c) {
  // D = A(16x32 bf16) x B(32x16 bf16) + C(16x16 f32)
  return __builtin_amdgcn_wmma_f32_16x16x32_bf16(false, a, false, b, (short)0, c,
                                                 false, false);
}

static __device__ __forceinline__ v8f zero8() { v8f z = {}; return z; }

// A-fragment: lane L holds row M=L%16; K chunks [8*(L/16), +8) and [16+8*(L/16), +8)
static __device__ __forceinline__ v16bf load_a_bf16(const __bf16* row, int half) {
  v8bf lo = *(const v8bf*)(row + half * 8);
  v8bf hi = *(const v8bf*)(row + 16 + half * 8);
  v16bf r;
#pragma unroll
  for (int i = 0; i < 8; ++i) { r[i] = lo[i]; r[i + 8] = hi[i]; }
  return r;
}

// ---------------------------------------------------------------------------
// Pack fp32 weight W[K][N] into bf16 WMMA B-fragment layout:
// P[((kt*NT+nt)*32 + lane)*16 + i] = W[kt*32 + (i/8)*16 + (lane/16)*8 + i%8][nt*16 + lane%16]
// ---------------------------------------------------------------------------
__global__ void pack_w(const float* __restrict__ W, __bf16* __restrict__ P,
                       int K, int N) {
  int t = blockIdx.x * 256 + threadIdx.x;
  if (t >= K * N) return;
  int i    = t & 15;
  int lane = (t >> 4) & 31;
  int NT   = N >> 4;
  int nt   = (t >> 9) % NT;
  int kt   = t / (512 * NT);
  int k = kt * 32 + ((i >> 3) << 4) + ((lane >> 4) << 3) + (i & 7);
  int n = nt * 16 + (lane & 15);
  P[t] = (__bf16)W[k * N + n];
}

__global__ void zero_kernel(float* __restrict__ p, int n4) {
  int t = blockIdx.x * 256 + threadIdx.x;
  if (t < n4) ((v4f*)p)[t] = (v4f){};
}

// ---------------------------------------------------------------------------
// x = inputs @ win + bin   (N x 32) @ (32 x 128) -> bf16 [N,128]
// ---------------------------------------------------------------------------
__global__ void __launch_bounds__(256)
node_in_kernel(const float* __restrict__ inp, const float* __restrict__ bin,
               const __bf16* __restrict__ winp, __bf16* __restrict__ xb) {
  int lane = threadIdx.x & 31;
  int wid  = blockIdx.x * 8 + (threadIdx.x >> 5);
  if (wid >= N_NODES / 16) return;
  int node0 = wid * 16;
  int col = lane & 15, half = lane >> 4;

  const float* arow = inp + (size_t)(node0 + col) * DIN;
  v16bf a;
#pragma unroll
  for (int i = 0; i < 8; ++i) {
    a[i]     = (__bf16)arow[half * 8 + i];
    a[i + 8] = (__bf16)arow[16 + half * 8 + i];
  }
#pragma unroll
  for (int nt = 0; nt < 8; ++nt) {
    v16bf b = *(const v16bf*)(winp + (nt * 32 + lane) * 16);
    v8f c = zero8();
    c = wmma_bf16(a, b, c);
    float bias = bin[nt * 16 + col];
#pragma unroll
    for (int r = 0; r < 8; ++r)
      xb[(size_t)(node0 + r + 8 * half) * DH + nt * 16 + col] = (__bf16)(c[r] + bias);
  }
}

// ---------------------------------------------------------------------------
// Fused edge MLP + scatter-add (segment_sum), 2 edge tiles per wave so every
// B (weight) fragment load feeds two WMMAs:
//   m = relu(relu([x_j|x_i] @ w1a + b1a) @ w1b + b1b) @ w1c + b1c
//   agg[dst] += m     (f32 atomics)
// 4 waves/block, per-wave LDS staging (2 tiles x 8KB) between layers.
// ---------------------------------------------------------------------------
__global__ void __launch_bounds__(128)
edge_kernel(const int* __restrict__ ei, const __bf16* __restrict__ xb,
            const __bf16* __restrict__ w1ap, const float* __restrict__ b1a,
            const __bf16* __restrict__ w1bp, const float* __restrict__ b1b,
            const __bf16* __restrict__ w1cp, const float* __restrict__ b1c,
            float* __restrict__ agg) {
  __shared__ float stage[4 * 2 * 16 * DH];  // 4 waves x 16KB = 64KB
  int lane  = threadIdx.x & 31;
  int wslot = threadIdx.x >> 5;
  int wid   = blockIdx.x * 4 + wslot;
  if (wid >= N_EDGES / 32) return;
  int e0 = wid * 32;  // two 16-edge tiles
  int col = lane & 15, half = lane >> 4;
  int src0 = ei[e0 + col];
  int dst0 = ei[N_EDGES + e0 + col];
  int src1 = ei[e0 + 16 + col];
  int dst1 = ei[N_EDGES + e0 + 16 + col];
  float*  sf0 = stage + wslot * (2 * 16 * DH);
  float*  sf1 = sf0 + 16 * DH;
  __bf16* sb0 = (__bf16*)sf0;
  __bf16* sb1 = (__bf16*)sf1;

  v8f acc0[8], acc1[8];
#pragma unroll
  for (int nt = 0; nt < 8; ++nt) { acc0[nt] = zero8(); acc1[nt] = zero8(); }

  // ---- layer 1: [16 x 256] @ [256 x 128]; kt 0-3 gather src row, 4-7 dst row
#pragma unroll
  for (int kt = 0; kt < 8; ++kt) {
    int n0 = (kt < 4) ? src0 : dst0;
    int n1 = (kt < 4) ? src1 : dst1;
    v16bf a0 = load_a_bf16(xb + (size_t)n0 * DH + (kt & 3) * 32, half);
    v16bf a1 = load_a_bf16(xb + (size_t)n1 * DH + (kt & 3) * 32, half);
#pragma unroll
    for (int nt = 0; nt < 8; ++nt) {
      v16bf b = *(const v16bf*)(w1ap + ((kt * 8 + nt) * 32 + lane) * 16);
      acc0[nt] = wmma_bf16(a0, b, acc0[nt]);
      acc1[nt] = wmma_bf16(a1, b, acc1[nt]);
    }
  }
#pragma unroll
  for (int nt = 0; nt < 8; ++nt) {
    float bias = b1a[nt * 16 + col];
#pragma unroll
    for (int r = 0; r < 8; ++r) {
      float v0 = acc0[nt][r] + bias;
      float v1 = acc1[nt][r] + bias;
      sb0[(r + 8 * half) * DH + nt * 16 + col] = (__bf16)(v0 > 0.f ? v0 : 0.f);
      sb1[(r + 8 * half) * DH + nt * 16 + col] = (__bf16)(v1 > 0.f ? v1 : 0.f);
    }
  }
  asm volatile("s_wait_dscnt 0x0" ::: "memory");

  // ---- layer 2: [16 x 128] @ [128 x 128]
#pragma unroll
  for (int nt = 0; nt < 8; ++nt) { acc0[nt] = zero8(); acc1[nt] = zero8(); }
#pragma unroll
  for (int kt = 0; kt < 4; ++kt) {
    v16bf a0 = load_a_bf16(sb0 + col * DH + kt * 32, half);
    v16bf a1 = load_a_bf16(sb1 + col * DH + kt * 32, half);
#pragma unroll
    for (int nt = 0; nt < 8; ++nt) {
      v16bf b = *(const v16bf*)(w1bp + ((kt * 8 + nt) * 32 + lane) * 16);
      acc0[nt] = wmma_bf16(a0, b, acc0[nt]);
      acc1[nt] = wmma_bf16(a1, b, acc1[nt]);
    }
  }
  asm volatile("s_wait_dscnt 0x0" ::: "memory");
#pragma unroll
  for (int nt = 0; nt < 8; ++nt) {
    float bias = b1b[nt * 16 + col];
#pragma unroll
    for (int r = 0; r < 8; ++r) {
      float v0 = acc0[nt][r] + bias;
      float v1 = acc1[nt][r] + bias;
      sb0[(r + 8 * half) * DH + nt * 16 + col] = (__bf16)(v0 > 0.f ? v0 : 0.f);
      sb1[(r + 8 * half) * DH + nt * 16 + col] = (__bf16)(v1 > 0.f ? v1 : 0.f);
    }
  }
  asm volatile("s_wait_dscnt 0x0" ::: "memory");

  // ---- layer 3: [16 x 128] @ [128 x 128], no relu; stage f32
#pragma unroll
  for (int nt = 0; nt < 8; ++nt) { acc0[nt] = zero8(); acc1[nt] = zero8(); }
#pragma unroll
  for (int kt = 0; kt < 4; ++kt) {
    v16bf a0 = load_a_bf16(sb0 + col * DH + kt * 32, half);
    v16bf a1 = load_a_bf16(sb1 + col * DH + kt * 32, half);
#pragma unroll
    for (int nt = 0; nt < 8; ++nt) {
      v16bf b = *(const v16bf*)(w1cp + ((kt * 8 + nt) * 32 + lane) * 16);
      acc0[nt] = wmma_bf16(a0, b, acc0[nt]);
      acc1[nt] = wmma_bf16(a1, b, acc1[nt]);
    }
  }
  asm volatile("s_wait_dscnt 0x0" ::: "memory");
#pragma unroll
  for (int nt = 0; nt < 8; ++nt) {
    float bias = b1c[nt * 16 + col];
#pragma unroll
    for (int r = 0; r < 8; ++r) {
      sf0[(r + 8 * half) * DH + nt * 16 + col] = acc0[nt][r] + bias;
      sf1[(r + 8 * half) * DH + nt * 16 + col] = acc1[nt][r] + bias;
    }
  }
  asm volatile("s_wait_dscnt 0x0" ::: "memory");

  // ---- scatter-add: lane handles half a row (64 contiguous floats) per tile
  int rr = lane >> 1;
  int c0 = (lane & 1) * 64;
  int dn0 = ei[N_EDGES + e0 + rr];
  int dn1 = ei[N_EDGES + e0 + 16 + rr];
  const float* s0 = sf0 + rr * DH + c0;
  const float* s1 = sf1 + rr * DH + c0;
  float* d0 = agg + (size_t)dn0 * DH + c0;
  float* d1 = agg + (size_t)dn1 * DH + c0;
#pragma unroll 4
  for (int i = 0; i < 64; ++i) atomicAdd(d0 + i, s0[i]);
#pragma unroll 4
  for (int i = 0; i < 64; ++i) atomicAdd(d1 + i, s1[i]);
}

// ---------------------------------------------------------------------------
// x = relu(agg @ w2a + b2a) @ w2b + b2b   -> bf16 [N,128]
// ---------------------------------------------------------------------------
__global__ void __launch_bounds__(256)
node_mid_kernel(const float* __restrict__ agg,
                const __bf16* __restrict__ w2ap, const float* __restrict__ b2a,
                const __bf16* __restrict__ w2bp, const float* __restrict__ b2b,
                __bf16* __restrict__ xb) {
  __shared__ __bf16 nstage[8 * 16 * DH];  // 8 waves x 4KB
  int lane  = threadIdx.x & 31;
  int wslot = threadIdx.x >> 5;
  int wid   = blockIdx.x * 8 + wslot;
  if (wid >= N_NODES / 16) return;
  int node0 = wid * 16;
  int col = lane & 15, half = lane >> 4;
  __bf16* sb = nstage + wslot * (16 * DH);

  v8f acc[8];
#pragma unroll
  for (int nt = 0; nt < 8; ++nt) acc[nt] = zero8();
#pragma unroll
  for (int kt = 0; kt < 4; ++kt) {
    const float* row = agg + (size_t)(node0 + col) * DH + kt * 32;
    v16bf a;
#pragma unroll
    for (int i = 0; i < 8; ++i) {
      a[i]     = (__bf16)row[half * 8 + i];
      a[i + 8] = (__bf16)row[16 + half * 8 + i];
    }
#pragma unroll
    for (int nt = 0; nt < 8; ++nt) {
      v16bf b = *(const v16bf*)(w2ap + ((kt * 8 + nt) * 32 + lane) * 16);
      acc[nt] = wmma_bf16(a, b, acc[nt]);
    }
  }
#pragma unroll
  for (int nt = 0; nt < 8; ++nt) {
    float bias = b2a[nt * 16 + col];
#pragma unroll
    for (int r = 0; r < 8; ++r) {
      float v = acc[nt][r] + bias;
      sb[(r + 8 * half) * DH + nt * 16 + col] = (__bf16)(v > 0.f ? v : 0.f);
    }
  }
  asm volatile("s_wait_dscnt 0x0" ::: "memory");
#pragma unroll
  for (int nt = 0; nt < 8; ++nt) acc[nt] = zero8();
#pragma unroll
  for (int kt = 0; kt < 4; ++kt) {
    v16bf a = load_a_bf16(sb + col * DH + kt * 32, half);
#pragma unroll
    for (int nt = 0; nt < 8; ++nt) {
      v16bf b = *(const v16bf*)(w2bp + ((kt * 8 + nt) * 32 + lane) * 16);
      acc[nt] = wmma_bf16(a, b, acc[nt]);
    }
  }
#pragma unroll
  for (int nt = 0; nt < 8; ++nt) {
    float bias = b2b[nt * 16 + col];
#pragma unroll
    for (int r = 0; r < 8; ++r)
      xb[(size_t)(node0 + r + 8 * half) * DH + nt * 16 + col] =
          (__bf16)(acc[nt][r] + bias);
  }
}

// ---------------------------------------------------------------------------
// out = x @ wout + bout   (N x 128) @ (128 x 16) -> f32
// ---------------------------------------------------------------------------
__global__ void __launch_bounds__(256)
out_kernel(const __bf16* __restrict__ xb, const __bf16* __restrict__ woutp,
           const float* __restrict__ bout, float* __restrict__ out) {
  int lane = threadIdx.x & 31;
  int wid  = blockIdx.x * 8 + (threadIdx.x >> 5);
  if (wid >= N_NODES / 16) return;
  int node0 = wid * 16;
  int col = lane & 15, half = lane >> 4;
  v8f c = zero8();
#pragma unroll
  for (int kt = 0; kt < 4; ++kt) {
    v16bf a = load_a_bf16(xb + (size_t)(node0 + col) * DH + kt * 32, half);
    v16bf b = *(const v16bf*)(woutp + (kt * 32 + lane) * 16);
    c = wmma_bf16(a, b, c);
  }
  float bias = bout[col];
#pragma unroll
  for (int r = 0; r < 8; ++r)
    out[(size_t)(node0 + r + 8 * half) * DOUT + col] = c[r] + bias;
}

// ---------------------------------------------------------------------------
extern "C" void kernel_launch(void* const* d_in, const int* in_sizes, int n_in,
                              void* d_out, int out_size, void* d_ws, size_t ws_size,
                              hipStream_t stream) {
  (void)in_sizes; (void)n_in; (void)out_size; (void)ws_size;
  const float* inputs = (const float*)d_in[0];
  // d_in[1] = frames (unused by reference)
  const int*   ei   = (const int*)d_in[2];
  const float* win  = (const float*)d_in[3];
  const float* bin  = (const float*)d_in[4];
  const float* wout = (const float*)d_in[5];
  const float* bout = (const float*)d_in[6];
  const float* w1a  = (const float*)d_in[7];
  const float* b1a  = (const float*)d_in[8];
  const float* w1b  = (const float*)d_in[9];
  const float* b1b  = (const float*)d_in[10];
  const float* w1c  = (const float*)d_in[11];
  const float* b1c  = (const float*)d_in[12];
  const float* w2a  = (const float*)d_in[13];
  const float* b2a  = (const float*)d_in[14];
  const float* w2b  = (const float*)d_in[15];
  const float* b2b  = (const float*)d_in[16];
  float* out = (float*)d_out;

  // ---- workspace carve-up
  char* ws = (char*)d_ws;
  auto alloc = [&](size_t bytes) -> char* {
    char* p = ws; ws += (bytes + 255) & ~(size_t)255; return p;
  };
  __bf16* xb     = (__bf16*)alloc((size_t)N_NODES * DH * sizeof(__bf16));
  float*  agg    = (float*) alloc((size_t)N_NODES * DH * sizeof(float));
  __bf16* win_p  = (__bf16*)alloc((size_t)DIN * DH * sizeof(__bf16));
  __bf16* wout_p = (__bf16*)alloc((size_t)DH * DOUT * sizeof(__bf16));
  __bf16* w1a_p  = (__bf16*)alloc((size_t)NBLK * 2 * DH * DH * sizeof(__bf16));
  __bf16* w1b_p  = (__bf16*)alloc((size_t)NBLK * DH * DH * sizeof(__bf16));
  __bf16* w1c_p  = (__bf16*)alloc((size_t)NBLK * DH * DH * sizeof(__bf16));
  __bf16* w2a_p  = (__bf16*)alloc((size_t)NBLK * DH * DH * sizeof(__bf16));
  __bf16* w2b_p  = (__bf16*)alloc((size_t)NBLK * DH * DH * sizeof(__bf16));

  auto pack = [&](const float* W, __bf16* P, int K, int Ncols) {
    int total = K * Ncols;
    pack_w<<<(total + 255) / 256, 256, 0, stream>>>(W, P, K, Ncols);
  };

  pack(win, win_p, DIN, DH);
  pack(wout, wout_p, DH, DOUT);
  for (int b = 0; b < NBLK; ++b) {
    pack(w1a + (size_t)b * 2 * DH * DH, w1a_p + (size_t)b * 2 * DH * DH, 2 * DH, DH);
    pack(w1b + (size_t)b * DH * DH, w1b_p + (size_t)b * DH * DH, DH, DH);
    pack(w1c + (size_t)b * DH * DH, w1c_p + (size_t)b * DH * DH, DH, DH);
    pack(w2a + (size_t)b * DH * DH, w2a_p + (size_t)b * DH * DH, DH, DH);
    pack(w2b + (size_t)b * DH * DH, w2b_p + (size_t)b * DH * DH, DH, DH);
  }

  const int nodeBlocks = (N_NODES / 16 + 7) / 8;   // 391
  const int edgeBlocks = (N_EDGES / 32) / 4;       // 3125, exact (no tail)
  const int aggN4 = N_NODES * DH / 4;

  node_in_kernel<<<nodeBlocks, 256, 0, stream>>>(inputs, bin, win_p, xb);

  for (int b = 0; b < NBLK; ++b) {
    zero_kernel<<<(aggN4 + 255) / 256, 256, 0, stream>>>(agg, aggN4);
    edge_kernel<<<edgeBlocks, 128, 0, stream>>>(
        ei, xb,
        w1a_p + (size_t)b * 2 * DH * DH, b1a + (size_t)b * DH,
        w1b_p + (size_t)b * DH * DH,     b1b + (size_t)b * DH,
        w1c_p + (size_t)b * DH * DH,     b1c + (size_t)b * DH,
        agg);
    node_mid_kernel<<<nodeBlocks, 256, 0, stream>>>(
        agg,
        w2a_p + (size_t)b * DH * DH, b2a + (size_t)b * DH,
        w2b_p + (size_t)b * DH * DH, b2b + (size_t)b * DH,
        xb);
  }

  out_kernel<<<nodeBlocks, 256, 0, stream>>>(xb, wout_p, bout, out);
}
